// SynapticAttention_47648367182496
// MI455X (gfx1250) — matrix operand order repaired and checked by reading, not compiled, not run on hardware
//
#include <hip/hip_runtime.h>

// ---------------------------------------------------------------------------
// Problem constants (from reference)
// ---------------------------------------------------------------------------
#define TT   12          // T
#define NN   4096        // N nodes
#define EE   65536       // E edges
#define CC   128         // HID
#define HH   8           // HEADS
#define HD   16          // head dim
#define HOR_ 12
#define IN_  16
#define SLOPE_ 0.2f

typedef __attribute__((ext_vector_type(16))) _Float16 v16h;
typedef __attribute__((ext_vector_type(8)))  _Float16 v8h;
typedef __attribute__((ext_vector_type(8)))  float    v8f;

// epilogue flag bits
#define GF_RELU  1
#define GF_ACCUM 2
#define GF_BIAS  4

// ---------------------------------------------------------------------------
// f16 WMMA GEMM:  C[M,N] (f32) = A[M,K] @ B^T  (+bias) (ReLU) (+=)
//   A row-major (lda, in halves).  B stored TRANSPOSED: B[n, k] row-major
//   (ldb, in halves) so every fragment load is one contiguous 32B v16h.
//   Batched over blockIdx.z with strides on B and C (A shared).
//   Block = 256 threads = 8 waves; block tile = 128 rows x (NT*16) cols;
//   wave tile = 16 rows x NT*16 cols (NT wmma tiles, A fragment reused NT x).
//   FLAGS is compile-time: straight-line epilogue, no runtime branches.
// ---------------------------------------------------------------------------
template <int NT, int FLAGS>
__global__ __launch_bounds__(256)
void k_gemm(const _Float16* __restrict__ A, const _Float16* __restrict__ B,
            float* __restrict__ C, const float* __restrict__ bias,
            int M, int K, int lda, int ldb, int ldc,
            long strideB, long strideC)
{
  B += (size_t)blockIdx.z * (size_t)strideB;
  C += (size_t)blockIdx.z * (size_t)strideC;

  const int wave = threadIdx.x >> 5;
  const int lane = threadIdx.x & 31;
  const int row0 = blockIdx.y * 128 + wave * 16;
  const int colBase = blockIdx.x * (NT * 16);

  const int hi  = lane >> 4;   // lane half: 0 or 1
  const int lid = lane & 15;

  v8f acc[NT];
  #pragma unroll
  for (int jt = 0; jt < NT; ++jt)
    #pragma unroll
    for (int q = 0; q < 8; ++q) acc[jt][q] = 0.0f;

  // A fragment: lanes 0-15 carry K k0+0..15, lanes 16-31 k0+16..31
  const _Float16* arow = A + (size_t)(row0 + lid) * lda + hi * 16;
  // B fragment: lane `lid` owns math-column colBase+jt*16+lid; lanes 16-31
  // take the K+16 half. One aligned 32B load per tile per k-step.
  const _Float16* brow = B + (size_t)(colBase + lid) * ldb + hi * 16;

  for (int k0 = 0; k0 < K; k0 += 32) {
    // prefetch the A stream ~8 k-steps ahead (speculative; OOB is dropped).
    __builtin_prefetch(arow + k0 + 256, 0, 3);
    v8h alo = *(const v8h*)(arow + k0);
    v8h ahi = *(const v8h*)(arow + k0 + 8);
    v16h a = __builtin_shufflevector(alo, ahi, 0, 1, 2, 3, 4, 5, 6, 7,
                                     8, 9, 10, 11, 12, 13, 14, 15);
    #pragma unroll
    for (int jt = 0; jt < NT; ++jt) {
      v16h b = *(const v16h*)(brow + (size_t)jt * 16 * ldb + k0);
      acc[jt] = __builtin_amdgcn_wmma_f32_16x16x32_f16(
          /*neg_a=*/false, a, /*neg_b=*/false, b,
          /*c_mod=*/(short)0, acc[jt], /*reuse_a=*/false, /*reuse_b=*/false);
    }
  }

  // C/D layout: VGPR i -> row (hi?8:0)+i, column = lane&15
  const int rbase = row0 + hi * 8;
  #pragma unroll
  for (int jt = 0; jt < NT; ++jt) {
    const int col = colBase + jt * 16 + lid;
    const float bv = (FLAGS & GF_BIAS) ? bias[col] : 0.0f;
    #pragma unroll
    for (int i = 0; i < 8; ++i) {
      float v = acc[jt][i] + bv;
      if (FLAGS & GF_RELU) v = fmaxf(v, 0.0f);
      const size_t idx = (size_t)(rbase + i) * ldc + col;
      if (FLAGS & GF_ACCUM) C[idx] += v; else C[idx] = v;
    }
  }
}

// ---------------------------------------------------------------------------
// Small utility kernels
// ---------------------------------------------------------------------------
__global__ void k_fill(float* __restrict__ p, float v, size_t n) {
  size_t i = (size_t)blockIdx.x * blockDim.x + threadIdx.x;
  if (i < n) p[i] = v;
}

__global__ void k_cvt(const float* __restrict__ s, _Float16* __restrict__ d, size_t n) {
  size_t i = (size_t)blockIdx.x * blockDim.x + threadIdx.x;
  if (i < n) d[i] = (_Float16)s[i];
}

// pack f32 [rows,cols] into f16 [rows,ldd] at column offset `off` (node-major)
__global__ void k_cvt_pack(const float* __restrict__ s, _Float16* __restrict__ d,
                           size_t rows, int cols, int ldd, int off) {
  size_t i = (size_t)blockIdx.x * blockDim.x + threadIdx.x;
  if (i < rows * (size_t)cols) {
    size_t r = i / cols; int c = (int)(i % cols);
    d[r * (size_t)ldd + off + c] = (_Float16)s[i];
  }
}

// batched transpose-convert: s[b, r, c] (f32) -> d[b, chanOff + c, r] (f16),
// dst batch stride = chanTot * rows.
__global__ void k_transpose16(const float* __restrict__ s, _Float16* __restrict__ d,
                              size_t total, int rows, int cols, int chanOff, int chanTot) {
  size_t i = (size_t)blockIdx.x * blockDim.x + threadIdx.x;
  if (i >= total) return;
  const int c = (int)(i % cols);
  const size_t rb = i / cols;
  const int r = (int)(rb % rows);
  const size_t b = rb / rows;
  d[b * (size_t)chanTot * rows + (size_t)(chanOff + c) * rows + r] = (_Float16)s[i];
}

// relu + row softmax over [NN] per row (relu happens before softmax, max >= 0)
__global__ __launch_bounds__(256)
void k_softmax_rows(float* __restrict__ adj, int n) {
  __shared__ float red[256];
  float* r = adj + (size_t)blockIdx.x * n;
  const int tid = threadIdx.x;
  float m = 0.0f;
  for (int j = tid; j < n; j += 256) m = fmaxf(m, fmaxf(r[j], 0.0f));
  red[tid] = m; __syncthreads();
  for (int s = 128; s > 0; s >>= 1) {
    if (tid < s) red[tid] = fmaxf(red[tid], red[tid + s]);
    __syncthreads();
  }
  m = red[0]; __syncthreads();
  float sum = 0.0f;
  for (int j = tid; j < n; j += 256) sum += __expf(fmaxf(r[j], 0.0f) - m);
  red[tid] = sum; __syncthreads();
  for (int s = 128; s > 0; s >>= 1) {
    if (tid < s) red[tid] += red[tid + s];
    __syncthreads();
  }
  const float inv = 1.0f / red[0];
  for (int j = tid; j < n; j += 256) r[j] = __expf(fmaxf(r[j], 0.0f) - m) * inv;
}

// h = x @ enc_W + enc_b     (x: [T*N,16], W: [16,128])
__global__ void k_encode(const float* __restrict__ x, const float* __restrict__ W,
                         const float* __restrict__ b, float* __restrict__ h) {
  size_t i = (size_t)blockIdx.x * blockDim.x + threadIdx.x;
  if (i >= (size_t)TT * NN * CC) return;
  const int c = (int)(i & (CC - 1));
  const size_t nt = i >> 7;
  const float* xr = x + nt * IN_;
  float acc = b[c];
  #pragma unroll
  for (int j = 0; j < IN_; ++j) acc += xr[j] * W[j * CC + c];
  h[i] = acc;
}

// s[t,n,h] = dot(xp[t,n,h,:], a_src[h]);  d likewise
__global__ void k_heads_dot(const float* __restrict__ xp, const float* __restrict__ As,
                            const float* __restrict__ Ad, float* __restrict__ sb,
                            float* __restrict__ db) {
  size_t i = (size_t)blockIdx.x * blockDim.x + threadIdx.x;
  if (i >= (size_t)TT * NN * HH) return;
  const int h = (int)(i & (HH - 1));
  const size_t nt = i >> 3;
  const float* xb = xp + nt * CC + h * HD;
  float s = 0.0f, d = 0.0f;
  #pragma unroll
  for (int j = 0; j < HD; ++j) { s += xb[j] * As[h * HD + j]; d += xb[j] * Ad[h * HD + j]; }
  sb[i] = s; db[i] = d;
}

// c[h] = dot(We[h*16..], a_edge[h])
__global__ void k_edge_const(const float* __restrict__ We, const float* __restrict__ Ae,
                             float* __restrict__ cb) {
  const int h = threadIdx.x;
  if (h < HH) {
    float c = 0.0f;
    #pragma unroll
    for (int j = 0; j < HD; ++j) c += We[h * HD + j] * Ae[h * HD + j];
    cb[h] = c;
  }
}

__device__ __forceinline__ void atomicMaxF(float* addr, float v) {
  if (v >= 0.0f) atomicMax((int*)addr, __float_as_int(v));
  else           atomicMin((unsigned int*)addr, __float_as_uint(v));
}

// per (t,e): leaky-relu logits for 8 heads; store; atomic segment-max on dst
__global__ void k_edge_logits(const float* __restrict__ sb, const float* __restrict__ db,
                              const float* __restrict__ ew, const int* __restrict__ ei,
                              const float* __restrict__ cb, float* __restrict__ lb,
                              float* __restrict__ mb) {
  size_t i = (size_t)blockIdx.x * blockDim.x + threadIdx.x;
  if (i >= (size_t)TT * EE) return;
  const int e = (int)(i % EE); const int t = (int)(i / EE);
  const int sN = ei[e], dN = ei[EE + e];
  const float w = ew[e];
  const float* sp = sb + ((size_t)t * NN + sN) * HH;
  const float* dp = db + ((size_t)t * NN + dN) * HH;
  float* lp = lb + i * HH;
  float* mp = mb + ((size_t)t * NN + dN) * HH;
  #pragma unroll
  for (int h = 0; h < HH; ++h) {
    float l = sp[h] + dp[h] + w * cb[h];
    l = (l > 0.0f) ? l : SLOPE_ * l;
    lp[h] = l;
    atomicMaxF(&mp[h], l);
  }
}

// per (t,e): ex = exp(l - m[dst]); store in-place; atomic segment-sum
__global__ void k_edge_exp(float* __restrict__ lb, const float* __restrict__ mb,
                           const int* __restrict__ ei, float* __restrict__ den) {
  size_t i = (size_t)blockIdx.x * blockDim.x + threadIdx.x;
  if (i >= (size_t)TT * EE) return;
  const int e = (int)(i % EE); const int t = (int)(i / EE);
  const int dN = ei[EE + e];
  float* lp = lb + i * HH;
  const float* mp = mb + ((size_t)t * NN + dN) * HH;
  float* dp = den + ((size_t)t * NN + dN) * HH;
  #pragma unroll
  for (int h = 0; h < HH; ++h) {
    const float ex = __expf(lp[h] - mp[h]);
    lp[h] = ex;
    atomicAdd(&dp[h], ex);
  }
}

// per (t,e): agg[t,dst,:] += alpha[h] * xp[t,src,:]
__global__ void k_edge_msg(const float* __restrict__ lb, const float* __restrict__ den,
                           const float* __restrict__ xp, const int* __restrict__ ei,
                           float* __restrict__ agg) {
  size_t i = (size_t)blockIdx.x * blockDim.x + threadIdx.x;
  if (i >= (size_t)TT * EE) return;
  const int e = (int)(i % EE); const int t = (int)(i / EE);
  const int sN = ei[e], dN = ei[EE + e];
  const float* lp = lb + i * HH;
  const float* dp = den + ((size_t)t * NN + dN) * HH;
  float a[HH];
  #pragma unroll
  for (int h = 0; h < HH; ++h) a[h] = lp[h] / (dp[h] + 1e-16f);
  const float* xs = xp + ((size_t)t * NN + sN) * CC;
  float* ag = agg + ((size_t)t * NN + dN) * CC;
  #pragma unroll 4
  for (int c = 0; c < CC; ++c) atomicAdd(&ag[c], a[c >> 4] * xs[c]);
}

__global__ void k_add_bias(float* __restrict__ p, const float* __restrict__ b, size_t n) {
  size_t i = (size_t)blockIdx.x * blockDim.x + threadIdx.x;
  if (i < n) p[i] += b[i & (CC - 1)];
}

// h_next = proc + conv + h_old
__global__ void k_hupdate(float* __restrict__ hn, const float* __restrict__ proc,
                          const float* __restrict__ cv, const float* __restrict__ ho,
                          size_t n) {
  size_t i = (size_t)blockIdx.x * blockDim.x + threadIdx.x;
  if (i < n) hn[i] = proc[i] + cv[i] + ho[i];
}

// rf = relu(outacc[t = T-1])
__global__ void k_relu_slice(const float* __restrict__ oa, float* __restrict__ rf) {
  size_t i = (size_t)blockIdx.x * blockDim.x + threadIdx.x;
  if (i < (size_t)NN * CC) rf[i] = fmaxf(oa[(size_t)(TT - 1) * NN * CC + i], 0.0f);
}

// y[n, HOR*IN] -> out[t, n, IN]
__global__ void k_final(const float* __restrict__ y, float* __restrict__ out) {
  size_t i = (size_t)blockIdx.x * blockDim.x + threadIdx.x;
  if (i >= (size_t)TT * NN * IN_) return;
  const int j = (int)(i & (IN_ - 1));
  const int n = (int)((i >> 4) & (NN - 1));
  const int t = (int)(i >> 16);
  out[i] = y[(size_t)n * (HOR_ * IN_) + t * IN_ + j];
}

// ---------------------------------------------------------------------------
// Host side
// ---------------------------------------------------------------------------
static inline unsigned cdiv(size_t a, unsigned b) { return (unsigned)((a + b - 1) / b); }

template <int NT, int FLAGS>
static void gemmT(const _Float16* A, const _Float16* B, float* C, const float* bias,
                  int M, int N, int K, int lda, int ldb, int ldc,
                  int batch, long sB, long sC, hipStream_t st) {
  dim3 g((unsigned)(N / (NT * 16)), (unsigned)(M / 128), (unsigned)batch);
  k_gemm<NT, FLAGS><<<g, 256, 0, st>>>(A, B, C, bias, M, K, lda, ldb, ldc, sB, sC);
}

extern "C" void kernel_launch(void* const* d_in, const int* in_sizes, int n_in,
                              void* d_out, int out_size, void* d_ws, size_t ws_size,
                              hipStream_t stream) {
  (void)in_sizes; (void)n_in; (void)out_size; (void)ws_size;

  const float* x     = (const float*)d_in[0];
  const int*   ei    = (const int*)  d_in[1];
  const float* ew    = (const float*)d_in[2];
  const float* encW  = (const float*)d_in[3];
  const float* encb  = (const float*)d_in[4];
  const float* semb  = (const float*)d_in[5];
  const float* temb  = (const float*)d_in[6];
  const float* attW  = (const float*)d_in[7];
  const float* asrc  = (const float*)d_in[8];
  const float* adst  = (const float*)d_in[9];
  const float* aedge = (const float*)d_in[10];
  const float* edgeW = (const float*)d_in[11];
  const float* attb  = (const float*)d_in[12];
  const float* convW = (const float*)d_in[13];
  const float* convb = (const float*)d_in[14];
  const float* skipW = (const float*)d_in[15];
  const float* skipb = (const float*)d_in[16];
  const float* roW1  = (const float*)d_in[17];
  const float* rob1  = (const float*)d_in[18];
  const float* roW2  = (const float*)d_in[19];
  const float* rob2  = (const float*)d_in[20];
  float* out = (float*)d_out;

  const size_t TNC = (size_t)TT * NN * CC;   // 6,291,456
  const size_t TNH = (size_t)TT * NN * HH;
  const size_t TE  = (size_t)TT * EE;

  // bump allocator over workspace
  char* wp = (char*)d_ws;
  auto alloc = [&](size_t bytes) -> void* {
    void* r = (void*)wp; wp += (bytes + 255) & ~(size_t)255; return r;
  };

  float*    adjf   = (float*)   alloc((size_t)NN * NN * 4);
  _Float16* adjh   = (_Float16*)alloc((size_t)NN * NN * 2);
  float*    hA     = (float*)   alloc(TNC * 4);
  float*    hB     = (float*)   alloc(TNC * 4);
  _Float16* hh     = (_Float16*)alloc(TNC * 2);        // node-major f16 h
  _Float16* hhT    = (_Float16*)alloc(TNC * 2);        // channel-major f16 h
  float*    xp     = (float*)   alloc(TNC * 4);        // reused as x1
  float*    lbuf   = (float*)   alloc(TE * HH * 4);    // reused as x2 (same size)
  float*    agg    = (float*)   alloc(TNC * 4);        // proc (f32)
  _Float16* proc16 = (_Float16*)alloc(TNC * 2);
  _Float16* cat    = (_Float16*)alloc(TNC * 2 * 2);    // [T*N, 256] node-major
  _Float16* catT   = (_Float16*)alloc(TNC * 2 * 2);    // [T, 256, N] channel-major
  float*    conv   = (float*)   alloc(TNC * 4);
  float*    outacc = (float*)   alloc(TNC * 4);
  float*    sbuf   = (float*)   alloc(TNH * 4);
  float*    dbuf   = (float*)   alloc(TNH * 4);
  float*    mbuf   = (float*)   alloc(TNH * 4);
  float*    den    = (float*)   alloc(TNH * 4);
  float*    cbuf   = (float*)   alloc(HH * 4);
  _Float16* w16    = (_Float16*)alloc((size_t)256 * 256 * 2); // transposed weights
  _Float16* src16  = (_Float16*)alloc((size_t)NN * CC * 2);
  _Float16* tgt16  = (_Float16*)alloc((size_t)NN * CC * 2);
  _Float16* rh     = (_Float16*)alloc((size_t)NN * 256 * 2);
  float*    r1f    = (float*)   alloc((size_t)NN * 256 * 4);
  float*    rf     = (float*)   alloc((size_t)NN * CC * 4);
  float*    yf     = (float*)   alloc((size_t)NN * 192 * 4);

  #define CVT(s, d, n)  k_cvt<<<cdiv((n), 256), 256, 0, stream>>>((s), (d), (n))
  #define FILL(p, v, n) k_fill<<<cdiv((n), 256), 256, 0, stream>>>((p), (v), (n))
  // transpose-convert weight W[K,N] (f32) -> WT[N,K] (f16) in w16
  #define WTX(Wp, Kr, Nc) k_transpose16<<<cdiv((size_t)(Kr) * (Nc), 256), 256, 0, stream>>>( \
      (Wp), w16, (size_t)(Kr) * (Nc), (Kr), (Nc), 0, (Nc))

  // ---- adjacency: softmax(relu(src_emb @ tgt_emb^T)) -----------------------
  CVT(semb, src16, (size_t)NN * CC);
  CVT(temb, tgt16, (size_t)NN * CC);   // tgt_emb is naturally [n][k] = B^T layout
  gemmT<8, 0>(src16, tgt16, adjf, nullptr, NN, NN, CC, CC, CC, NN, 1, 0, 0, stream);
  k_softmax_rows<<<NN, 256, 0, stream>>>(adjf, NN);
  CVT(adjf, adjh, (size_t)NN * NN);

  // ---- encoder -------------------------------------------------------------
  k_encode<<<cdiv(TNC, 256), 256, 0, stream>>>(x, encW, encb, hA);
  FILL(outacc, 0.0f, TNC);

  float* hcur = hA;
  float* hnext = hB;

  for (int i = 0; i < 2; ++i) {
    const float* Wi  = attW  + (size_t)i * CC * CC;
    const float* Asi = asrc  + (size_t)i * HH * HD;
    const float* Adi = adst  + (size_t)i * HH * HD;
    const float* Aei = aedge + (size_t)i * HH * HD;
    const float* Wei = edgeW + (size_t)i * CC;
    const float* bbi = attb  + (size_t)i * CC;
    const float* Wci = convW + (size_t)i * 2 * CC * CC;
    const float* bci = convb + (size_t)i * CC;
    const float* Wsi = skipW + (size_t)i * CC * CC;
    const float* bsi = skipb + (size_t)i * CC;

    // h -> f16 (node-major for A operands, channel-major for B operands)
    CVT(hcur, hh, TNC);
    k_transpose16<<<cdiv(TNC, 256), 256, 0, stream>>>(hcur, hhT, TNC, NN, CC, 0, CC);

    // xp = h @ attW[i]
    WTX(Wi, CC, CC);
    gemmT<8, 0>(hh, w16, xp, nullptr, TT * NN, CC, CC, CC, CC, CC, 1, 0, 0, stream);

    // GAT
    k_heads_dot<<<cdiv(TNH, 256), 256, 0, stream>>>(xp, Asi, Adi, sbuf, dbuf);
    k_edge_const<<<1, 32, 0, stream>>>(Wei, Aei, cbuf);
    FILL(mbuf, -3.0e38f, TNH);
    FILL(den, 0.0f, TNH);
    FILL(agg, 0.0f, TNC);
    k_edge_logits<<<cdiv(TE, 256), 256, 0, stream>>>(sbuf, dbuf, ew, ei, cbuf, lbuf, mbuf);
    k_edge_exp   <<<cdiv(TE, 256), 256, 0, stream>>>(lbuf, mbuf, ei, den);
    k_edge_msg   <<<cdiv(TE, 256), 256, 0, stream>>>(lbuf, den, xp, ei, agg);
    k_add_bias<<<cdiv(TNC, 256), 256, 0, stream>>>(agg, bbi, TNC);   // proc = agg + bias

    // out += proc @ skipW[i] + skipb[i]
    CVT(agg, proc16, TNC);
    WTX(Wsi, CC, CC);
    gemmT<8, GF_ACCUM | GF_BIAS>(proc16, w16, outacc, bsi, TT * NN, CC, CC,
                                 CC, CC, CC, 1, 0, 0, stream);

    // dense conv: x1 = adj@h ; x2 = adj@x1 ; conv = [x1|x2] @ convW + convb
    float* x1 = xp;    // xp no longer needed
    float* x2 = lbuf;  // lbuf no longer needed
    gemmT<8, 0>(adjh, hhT, x1, nullptr, NN, CC, NN, NN, NN, CC,
                /*batch=*/TT, (long)NN * CC, (long)NN * CC, stream);
    k_cvt_pack<<<cdiv(TNC, 256), 256, 0, stream>>>(x1, cat, (size_t)TT * NN, CC, 2 * CC, 0);
    k_transpose16<<<cdiv(TNC, 256), 256, 0, stream>>>(x1, catT, TNC, NN, CC, 0, 2 * CC);
    gemmT<8, 0>(adjh, catT, x2, nullptr, NN, CC, NN, NN, NN, CC,
                /*batch=*/TT, (long)NN * 2 * CC, (long)NN * CC, stream);
    k_cvt_pack<<<cdiv(TNC, 256), 256, 0, stream>>>(x2, cat, (size_t)TT * NN, CC, 2 * CC, CC);
    WTX(Wci, 2 * CC, CC);
    gemmT<8, GF_BIAS>(cat, w16, conv, bci, TT * NN, CC, 2 * CC,
                      2 * CC, 2 * CC, CC, 1, 0, 0, stream);

    // h = proc + conv + h_old
    k_hupdate<<<cdiv(TNC, 256), 256, 0, stream>>>(hnext, agg, conv, hcur, TNC);
    float* tmp = hcur; hcur = hnext; hnext = tmp;
  }

  // ---- readout -------------------------------------------------------------
  k_relu_slice<<<cdiv((size_t)NN * CC, 256), 256, 0, stream>>>(outacc, rf);
  CVT(rf, rh, (size_t)NN * CC);
  WTX(roW1, CC, 256);
  gemmT<16, GF_RELU | GF_BIAS>(rh, w16, r1f, rob1, NN, 256, CC,
                               CC, CC, 256, 1, 0, 0, stream);
  CVT(r1f, rh, (size_t)NN * 256);
  WTX(roW2, 256, 192);
  gemmT<12, GF_BIAS>(rh, w16, yf, rob2, NN, 192, 256, 256, 256, 192, 1, 0, 0, stream);
  k_final<<<cdiv((size_t)TT * NN * IN_, 256), 256, 0, stream>>>(yf, out);

  #undef CVT
  #undef FILL
  #undef WTX
}
